// DefaultEncoder_19980187861411
// MI455X (gfx1250) — matrix operand order repaired and checked by reading, not compiled, not run on hardware
//
#include <hip/hip_runtime.h>

typedef __bf16 bf16;
typedef float v8f __attribute__((ext_vector_type(8)));
typedef bf16 v16bf __attribute__((ext_vector_type(16)));

union FragU { v16bf v; unsigned int u[8]; };

// Each wave computes MSUB vertically-stacked 16x16 C tiles (one N column tile),
// sharing the B fragments across all MSUB accumulators.
// M must be a multiple of 16*MSUB (50000 = 625 * 80).
#define MSUB 5

// ---------------------------------------------------------------------------
// WMMA fragment loaders (wave32); A and B use the same contiguous k order, k = 16h + s.
// A (16x32 bf16, row-major MxK): lanes 0-15 = rows M=0..15 with
//   V0..7 -> K = 2v,2v+1 ; lanes 16-31 add +16 to K.
// B (32x16 bf16): lane = column n, VGPR v packs K=2v (lo), 2v+1 (hi) for
//   lanes 0-15; lanes 16-31 cover K=16..31.  We store weights TRANSPOSED
//   (N x K row-major), so this is a contiguous 32-byte per-lane read.
// C/D (16x16 f32): VGPR r, lane l -> C[r + 8*(l>>4)][l&15].
// ---------------------------------------------------------------------------
__device__ __forceinline__ void load_frag_A(FragU& f, const bf16* __restrict__ A,
                                            int K, int row, int k0, int hh) {
  const bf16* base = A + (size_t)row * K + k0 + 16 * hh;
#pragma unroll
  for (int v = 0; v < 4; ++v) f.u[v] = *(const unsigned int*)(base + 2 * v);
#pragma unroll
  for (int v = 4; v < 8; ++v) f.u[v] = *(const unsigned int*)(base + 2 * v);
}

// Bt is the transposed weight (N x K row-major).
__device__ __forceinline__ void load_frag_Bt(FragU& f, const bf16* __restrict__ Bt,
                                             int K, int col, int k0, int hh) {
  const bf16* base = Bt + (size_t)col * K + k0 + 16 * hh;
#pragma unroll
  for (int v = 0; v < 8; ++v) f.u[v] = *(const unsigned int*)(base + 2 * v);
}

// C = epilogue(A1@B1 [+ A2@B2] + bias), epilogue = PReLU(alpha) if alpha!=null.
// B1t/B2t are transposed (N x K). Writes f32 output (for the scatter phase)
// and optionally bf16 (next GEMM input). 8 waves (256 threads) per block,
// each wave owns an 80x16 strip of C. DUAL is compile-time so the K-loop is
// branch-free straight-line code.
template <bool DUAL>
__global__ void wmma_gemm_k(const bf16* __restrict__ A1, const bf16* __restrict__ B1t,
                            const bf16* __restrict__ A2, const bf16* __restrict__ B2t,
                            const float* __restrict__ bias, const float* __restrict__ alpha,
                            float* __restrict__ Cf, bf16* __restrict__ Cb,
                            int M, int N, int K) {
  int wave = blockIdx.x * (blockDim.x >> 5) + (threadIdx.x >> 5);
  int tiles_n = N >> 4;
  int mgroups = M / (16 * MSUB);
  if (wave >= mgroups * tiles_n) return;  // wave-uniform; EXEC stays all-ones
  int tg = wave / tiles_n;
  int tn = wave - tg * tiles_n;
  int lane = threadIdx.x & 31;
  int hh = lane >> 4;
  int l = lane & 15;
  int row0 = tg * (16 * MSUB) + l;  // A row (lane part) of subtile 0
  int col = tn * 16 + l;            // B/C column for this lane

  v8f acc[MSUB];
#pragma unroll
  for (int i = 0; i < MSUB; ++i) acc[i] = (v8f){};

  for (int k0 = 0; k0 < K; k0 += 32) {
    FragU b1;
    load_frag_Bt(b1, B1t, K, col, k0, hh);
    FragU b2;
    if constexpr (DUAL) load_frag_Bt(b2, B2t, K, col, k0, hh);

    // Pass 1: 5 independent accumulators -> independent WMMAs.
#pragma unroll
    for (int i = 0; i < MSUB; ++i) {
      FragU a;
      load_frag_A(a, A1, K, row0 + 16 * i, k0, hh);
      acc[i] = __builtin_amdgcn_wmma_f32_16x16x32_bf16(false, a.v, false, b1.v,
                                                       (short)0, acc[i], false, false);
    }
    // Pass 2 (dual input): each acc[i] sees 4 independent WMMAs between its
    // two updates, covering the WMMA->WMMA co-execution hazard window.
    if constexpr (DUAL) {
#pragma unroll
      for (int i = 0; i < MSUB; ++i) {
        FragU a;
        load_frag_A(a, A2, K, row0 + 16 * i, k0, hh);
        acc[i] = __builtin_amdgcn_wmma_f32_16x16x32_bf16(false, a.v, false, b2.v,
                                                         (short)0, acc[i], false, false);
      }
    }
  }

  float bn = bias[col];
  float an = alpha ? alpha[col] : 1.0f;
#pragma unroll
  for (int i = 0; i < MSUB; ++i) {
#pragma unroll
    for (int r = 0; r < 8; ++r) {
      int m = tg * (16 * MSUB) + 16 * i + r + 8 * hh;
      float val = acc[i][r] + bn;
      if (alpha) val = val > 0.0f ? val : an * val;
      size_t o = (size_t)m * N + col;
      Cf[o] = val;
      if (Cb) Cb[o] = (bf16)val;
    }
  }
}

// LayerNorm(256) + ReLU, writes bf16. One block (256 threads) per row.
__global__ void ln_relu_to_bf16(const float* __restrict__ X, const float* __restrict__ g,
                                const float* __restrict__ be, bf16* __restrict__ Y, int N) {
  __shared__ float red[256];
  int row = blockIdx.x;
  int t = threadIdx.x;
  float v = X[(size_t)row * N + t];
  red[t] = v;
  __syncthreads();
  for (int s = 128; s > 0; s >>= 1) {
    if (t < s) red[t] += red[t + s];
    __syncthreads();
  }
  float mu = red[0] / (float)N;
  __syncthreads();
  float d = v - mu;
  red[t] = d * d;
  __syncthreads();
  for (int s = 128; s > 0; s >>= 1) {
    if (t < s) red[t] += red[t + s];
    __syncthreads();
  }
  float var = red[0] / (float)N;
  float h = d * rsqrtf(var + 1e-5f) * g[t] + be[t];
  h = h > 0.0f ? h : 0.0f;
  Y[(size_t)row * N + t] = (bf16)h;
}

__global__ void zero_f32(float* __restrict__ p, long long n) {
  long long i = blockIdx.x * (long long)blockDim.x + threadIdx.x;
  long long stride = (long long)gridDim.x * blockDim.x;
  for (; i < n; i += stride) p[i] = 0.0f;
}

__global__ void f32_to_bf16(const float* __restrict__ in, bf16* __restrict__ out, long long n) {
  long long i = blockIdx.x * (long long)blockDim.x + threadIdx.x;
  long long stride = (long long)gridDim.x * blockDim.x;
  for (; i < n; i += stride) out[i] = (bf16)in[i];
}

// Convert f32 weight (K x N row-major) to bf16 TRANSPOSED (N x K row-major).
__global__ void f32_to_bf16_T(const float* __restrict__ in, bf16* __restrict__ out,
                              int K, int N) {
  long long total = (long long)K * N;
  long long i = blockIdx.x * (long long)blockDim.x + threadIdx.x;
  long long stride = (long long)gridDim.x * blockDim.x;
  for (; i < total; i += stride) {
    int k = (int)(i / N);
    int n = (int)(i - (long long)k * N);
    out[(size_t)n * K + k] = (bf16)in[i];
  }
}

// Per-edge degree count at destination nodes.
__global__ void degree_count(const int* __restrict__ dst, float* __restrict__ cnt, int E) {
  long long i = blockIdx.x * (long long)blockDim.x + threadIdx.x;
  long long stride = (long long)gridDim.x * blockDim.x;
  for (; i < E; i += stride) atomicAdd(&cnt[dst[i]], 1.0f);
}

// aggr[dst] += h[src], 4 channels per thread (float4 read, f32 atomics).
__global__ void scatter_add4(const float* __restrict__ H, const int* __restrict__ src,
                             const int* __restrict__ dst, float* __restrict__ aggr,
                             long long total, int d) {
  long long i = blockIdx.x * (long long)blockDim.x + threadIdx.x;
  long long stride = (long long)gridDim.x * blockDim.x;
  int c4 = d >> 2;
  for (; i < total; i += stride) {
    long long e = i / c4;
    int c = (int)(i - e * c4) * 4;
    int s = src[e];
    int t = dst[e];
    float4 v = *(const float4*)(H + (size_t)s * d + c);
    float* p = aggr + (size_t)t * d + c;
    atomicAdd(p + 0, v.x);
    atomicAdd(p + 1, v.y);
    atomicAdd(p + 2, v.z);
    atomicAdd(p + 3, v.w);
  }
}

// out = bf16(aggr / max(cnt,1))
__global__ void mean_to_bf16(const float* __restrict__ aggr, const float* __restrict__ cnt,
                             bf16* __restrict__ out, long long total, int d) {
  long long i = blockIdx.x * (long long)blockDim.x + threadIdx.x;
  long long stride = (long long)gridDim.x * blockDim.x;
  for (; i < total; i += stride) {
    long long row = i / d;
    out[i] = (bf16)(aggr[i] / fmaxf(cnt[row], 1.0f));
  }
}

__global__ void copy_out_f32(const float* __restrict__ in, float* __restrict__ out, int n) {
  int i = blockIdx.x * blockDim.x + threadIdx.x;
  if (i < n) out[i] = in[i];
}

extern "C" void kernel_launch(void* const* d_in, const int* in_sizes, int n_in,
                              void* d_out, int out_size, void* d_ws, size_t ws_size,
                              hipStream_t stream) {
  const float* x = (const float*)d_in[0];
  const int* ei = (const int*)d_in[1];
  const float* W1 = (const float*)d_in[3];
  const float* b1 = (const float*)d_in[4];
  const float* g1 = (const float*)d_in[5];
  const float* be1 = (const float*)d_in[6];
  const float* W2 = (const float*)d_in[7];
  const float* b2 = (const float*)d_in[8];
  const float* Wl1 = (const float*)d_in[9];
  const float* bl1 = (const float*)d_in[10];
  const float* Wr1 = (const float*)d_in[11];
  const float* Wl2 = (const float*)d_in[12];
  const float* bl2 = (const float*)d_in[13];
  const float* Wr2 = (const float*)d_in[14];
  const float* Wl3 = (const float*)d_in[15];
  const float* bl3 = (const float*)d_in[16];
  const float* Wr3 = (const float*)d_in[17];
  const float* a1 = (const float*)d_in[18];
  const float* a2 = (const float*)d_in[19];
  const float* a3 = (const float*)d_in[20];

  const int IN_CH = 128, HID = 256, HC = 128, D2 = 512;
  const int NN = in_sizes[0] / IN_CH;  // 50000 = 625 * 80
  const int E = in_sizes[1] / 2;       // 800000
  const int* src = ei;
  const int* dst = ei + E;

  // ---- workspace carving (256B aligned) ----
  char* ws = (char*)d_ws;
  size_t off = 0;
  auto carve = [&](size_t bytes) -> void* {
    void* p = ws + off;
    off = (off + bytes + 255) & ~(size_t)255;
    return p;
  };
  float* hF = (float*)carve((size_t)NN * D2 * sizeof(float));     // f32 features
  float* aggrF = (float*)carve((size_t)NN * D2 * sizeof(float));  // f32 aggregate
  bf16* hB0 = (bf16*)carve((size_t)NN * D2 * sizeof(bf16));
  bf16* hB1 = (bf16*)carve((size_t)NN * D2 * sizeof(bf16));
  bf16* aggrB = (bf16*)carve((size_t)NN * D2 * sizeof(bf16));
  float* cnt = (float*)carve((size_t)NN * sizeof(float));
  bf16* W1b = (bf16*)carve((size_t)IN_CH * HID * sizeof(bf16));   // all transposed (N x K)
  bf16* W2b = (bf16*)carve((size_t)HID * HC * sizeof(bf16));
  bf16* Wl1b = (bf16*)carve((size_t)HC * 256 * sizeof(bf16));
  bf16* Wr1b = (bf16*)carve((size_t)HC * 256 * sizeof(bf16));
  bf16* Wl2b = (bf16*)carve((size_t)256 * 512 * sizeof(bf16));
  bf16* Wr2b = (bf16*)carve((size_t)256 * 512 * sizeof(bf16));
  bf16* Wl3b = (bf16*)carve((size_t)512 * HC * sizeof(bf16));
  bf16* Wr3b = (bf16*)carve((size_t)512 * HC * sizeof(bf16));

  auto gsgrid = [](long long n) {
    long long b = (n + 255) / 256;
    if (b > 262144) b = 262144;
    return (int)(b > 0 ? b : 1);
  };
  auto gemm_grid = [](int M, int N) {
    int waves = (M / (16 * MSUB)) * (N >> 4);
    return (waves + 7) / 8;
  };

  // ---- convert weights (transposed) & input to bf16 ----
  f32_to_bf16_T<<<gsgrid((long long)IN_CH * HID), 256, 0, stream>>>(W1, W1b, IN_CH, HID);
  f32_to_bf16_T<<<gsgrid((long long)HID * HC), 256, 0, stream>>>(W2, W2b, HID, HC);
  f32_to_bf16_T<<<gsgrid((long long)HC * 256), 256, 0, stream>>>(Wl1, Wl1b, HC, 256);
  f32_to_bf16_T<<<gsgrid((long long)HC * 256), 256, 0, stream>>>(Wr1, Wr1b, HC, 256);
  f32_to_bf16_T<<<gsgrid((long long)256 * 512), 256, 0, stream>>>(Wl2, Wl2b, 256, 512);
  f32_to_bf16_T<<<gsgrid((long long)256 * 512), 256, 0, stream>>>(Wr2, Wr2b, 256, 512);
  f32_to_bf16_T<<<gsgrid((long long)512 * HC), 256, 0, stream>>>(Wl3, Wl3b, 512, HC);
  f32_to_bf16_T<<<gsgrid((long long)512 * HC), 256, 0, stream>>>(Wr3, Wr3b, 512, HC);
  f32_to_bf16<<<gsgrid((long long)NN * IN_CH), 256, 0, stream>>>(x, hB0, (long long)NN * IN_CH);

  // ---- encoder: GEMM1 -> LN+ReLU -> GEMM2 ----
  wmma_gemm_k<false><<<gemm_grid(NN, HID), 256, 0, stream>>>(
      hB0, W1b, nullptr, nullptr, b1, nullptr, aggrF, nullptr, NN, HID, IN_CH);
  ln_relu_to_bf16<<<NN, 256, 0, stream>>>(aggrF, g1, be1, hB1, HID);
  wmma_gemm_k<false><<<gemm_grid(NN, HC), 256, 0, stream>>>(
      hB1, W2b, nullptr, nullptr, b2, nullptr, hF, hB0, NN, HC, HID);

  // ---- degree (shared by all layers) ----
  zero_f32<<<gsgrid(NN), 256, 0, stream>>>(cnt, NN);
  degree_count<<<gsgrid(E), 256, 0, stream>>>(dst, cnt, E);

  // ---- SAGE layer 1: 128 -> 256 ----
  zero_f32<<<gsgrid((long long)NN * HC), 256, 0, stream>>>(aggrF, (long long)NN * HC);
  scatter_add4<<<gsgrid((long long)E * (HC / 4)), 256, 0, stream>>>(
      hF, src, dst, aggrF, (long long)E * (HC / 4), HC);
  mean_to_bf16<<<gsgrid((long long)NN * HC), 256, 0, stream>>>(
      aggrF, cnt, aggrB, (long long)NN * HC, HC);
  wmma_gemm_k<true><<<gemm_grid(NN, 256), 256, 0, stream>>>(
      aggrB, Wl1b, hB0, Wr1b, bl1, a1, hF, hB1, NN, 256, HC);

  // ---- SAGE layer 2: 256 -> 512 ----
  zero_f32<<<gsgrid((long long)NN * 256), 256, 0, stream>>>(aggrF, (long long)NN * 256);
  scatter_add4<<<gsgrid((long long)E * (256 / 4)), 256, 0, stream>>>(
      hF, src, dst, aggrF, (long long)E * (256 / 4), 256);
  mean_to_bf16<<<gsgrid((long long)NN * 256), 256, 0, stream>>>(
      aggrF, cnt, aggrB, (long long)NN * 256, 256);
  wmma_gemm_k<true><<<gemm_grid(NN, 512), 256, 0, stream>>>(
      aggrB, Wl2b, hB1, Wr2b, bl2, a2, hF, hB0, NN, 512, 256);

  // ---- SAGE layer 3: 512 -> 128 ----
  zero_f32<<<gsgrid((long long)NN * 512), 256, 0, stream>>>(aggrF, (long long)NN * 512);
  scatter_add4<<<gsgrid((long long)E * (512 / 4)), 256, 0, stream>>>(
      hF, src, dst, aggrF, (long long)E * (512 / 4), 512);
  mean_to_bf16<<<gsgrid((long long)NN * 512), 256, 0, stream>>>(
      aggrF, cnt, aggrB, (long long)NN * 512, 512);
  wmma_gemm_k<true><<<gemm_grid(NN, HC), 256, 0, stream>>>(
      aggrB, Wl3b, hB0, Wr3b, bl3, a3, hF, nullptr, NN, HC, 512);

  // ---- output: first batch_size(=out_size/HC) rows ----
  copy_out_f32<<<(out_size + 255) / 256, 256, 0, stream>>>(hF, (float*)d_out, out_size);
}
